// Drug_14070312861994
// MI455X (gfx1250) — matrix operand order repaired and checked by reading, not compiled, not run on hardware
//
#include <hip/hip_runtime.h>
#include <hip/hip_bf16.h>
#include <math.h>

typedef __attribute__((ext_vector_type(16))) _Float16 v16h;
typedef __attribute__((ext_vector_type(8)))  _Float16 h8;
typedef __attribute__((ext_vector_type(8)))  float    v8f;

#define CDIV(a,b) (((a)+(b)-1)/(b))

static constexpr int NATOMS = 8192;
static constexpr int NEDGE  = 32768;
static constexpr int NG     = 256;
static constexpr int DIN    = 78;
static constexpr int NH     = 10;
static constexpr int DH     = 780;    // DIN*NH
static constexpr int DFF    = 2048;
static constexpr int DFC1   = 1024;
static constexpr int DFC2   = 128;
static constexpr int NSELF  = NEDGE + NATOMS;   // edges + self loops

// ---------- order-preserving float<->uint key for atomicMax on floats ----------
__device__ __forceinline__ unsigned f2k(float f) {
  unsigned u = __float_as_uint(f);
  return (u & 0x80000000u) ? ~u : (u | 0x80000000u);
}
__device__ __forceinline__ float k2f(unsigned k) {
  unsigned u = (k & 0x80000000u) ? (k & 0x7fffffffu) : ~k;
  return __uint_as_float(u);
}

// =====================================================================
// WMMA GEMM on pre-converted f16 operands.
//   A : f16 [M,  Kd] row-major, lda  (Kd mult of 32, M mult of 64)
//   B : f16 [>=ceil128(N), Kd] row-major == B^T of the math (trans layout)
//   C = act(alpha * A @ B^T_rows + bias)
// Block tile 64x128, 128 threads (4 waves); each wave owns 32x64 =
// 2x4 v_wmma_f32_16x16x32_f16 per K-step (8 WMMA per 12 ds_load_b128).
// =====================================================================
template<bool OUT_HALF, bool RELU, bool BIAS>
__global__ __launch_bounds__(128)
void gemm_wmma(const _Float16* __restrict__ A, const _Float16* __restrict__ B,
               const float* __restrict__ bias, void* __restrict__ Cv,
               int M, int N, int Kd, int lda, int ldb, int ldc, float alpha)
{
  __shared__ _Float16 As[64][40];    // 64 rows  x 32 K (+pad; 80B row keeps 16B align)
  __shared__ _Float16 Bs[128][40];   // 128 cols x 32 K

  const int t    = threadIdx.x;
  const int lane = t & 31;
  const int wid  = t >> 5;
  const int mW   = (wid >> 1) * 32;
  const int nW   = (wid & 1) * 64;
  const size_t blockM = (size_t)blockIdx.y * 64;
  const size_t blockN = (size_t)blockIdx.x * 128;

  const int kb = (lane >> 4) * 8;    // A fragment half-wave K base
  const int kh = (lane >> 4) * 16;   // unused; B fragment also uses kb

  v8f acc[2][4] = {};

  for (int k0 = 0; k0 < Kd; k0 += 32) {
    // ---- stage tiles: unguarded 16B chunks (A: 2/thread, B: 4/thread) ----
#pragma unroll
    for (int i = 0; i < 2; ++i) {
      int chunk = i * 128 + t;
      int r  = chunk >> 2;
      int c8 = (chunk & 3) * 8;
      *(h8*)&As[r][c8] = *(const h8*)&A[(blockM + r) * lda + k0 + c8];
    }
#pragma unroll
    for (int i = 0; i < 4; ++i) {
      int chunk = i * 128 + t;
      int r  = chunk >> 2;
      int c8 = (chunk & 3) * 8;
      *(h8*)&Bs[r][c8] = *(const h8*)&B[(blockN + r) * ldb + k0 + c8];
    }
    __syncthreads();

    v16h afr[2], bfr[4];
#pragma unroll
    for (int mi = 0; mi < 2; ++mi) {
      int row = mW + mi * 16 + (lane & 15);
      h8 a0 = *(const h8*)&As[row][kb];
      h8 a1 = *(const h8*)&As[row][16 + kb];
      afr[mi] = __builtin_shufflevector(a0, a1, 0,1,2,3,4,5,6,7,8,9,10,11,12,13,14,15);
    }
#pragma unroll
    for (int ni = 0; ni < 4; ++ni) {
      int col = nW + ni * 16 + (lane & 15);
      h8 b0 = *(const h8*)&Bs[col][kb];
      h8 b1 = *(const h8*)&Bs[col][16 + kb];
      bfr[ni] = __builtin_shufflevector(b0, b1, 0,1,2,3,4,5,6,7,8,9,10,11,12,13,14,15);
    }
#pragma unroll
    for (int mi = 0; mi < 2; ++mi)
#pragma unroll
      for (int ni = 0; ni < 4; ++ni)
        acc[mi][ni] = __builtin_amdgcn_wmma_f32_16x16x32_f16(
            false, afr[mi], false, bfr[ni], (short)0, acc[mi][ni], false, false);
    __syncthreads();
  }

  // ---- epilogue ----
#pragma unroll
  for (int mi = 0; mi < 2; ++mi)
#pragma unroll
    for (int ni = 0; ni < 4; ++ni) {
      int n = (int)blockN + nW + ni * 16 + (lane & 15);
      if (n >= N) continue;
#pragma unroll
      for (int r = 0; r < 8; ++r) {
        size_t m = blockM + mW + mi * 16 + r + ((lane >> 4) << 3);
        float vout = acc[mi][ni][r] * alpha;
        if constexpr (BIAS) vout += bias[n];
        if constexpr (RELU) vout = vout > 0.0f ? vout : 0.0f;
        if constexpr (OUT_HALF) ((_Float16*)Cv)[m * ldc + n] = (_Float16)vout;
        else                    ((float*)Cv)[m * ldc + n] = vout;
      }
    }
}

// ---------------- fp32 -> f16 conversions (pad / transpose+pad) ----------------
// dst[r][c] (ldd=Cpad) = src[r][c], zero for c>=C
__global__ void k_cvtA(const float* __restrict__ src, _Float16* __restrict__ dst,
                       int R, int C, int Cpad) {
  int i = blockIdx.x * blockDim.x + threadIdx.x;
  if (i < R * Cpad) {
    int r = i / Cpad, c = i % Cpad;
    dst[i] = (c < C) ? (_Float16)src[(size_t)r * C + c] : (_Float16)0.0f;
  }
}
// src fp32 [R,C] -> dst f16 [Cpad][Rpad]: dst[c][r] = src[r][c], zero-padded
__global__ void k_cvtT(const float* __restrict__ src, _Float16* __restrict__ dst,
                       int R, int C, int Rpad, int Cpad) {
  int i = blockIdx.x * blockDim.x + threadIdx.x;
  if (i < Cpad * Rpad) {
    int c = i / Rpad, r = i % Rpad;
    dst[i] = (c < C && r < R) ? (_Float16)src[(size_t)r * C + c] : (_Float16)0.0f;
  }
}

// ---------------- softmax over one f16 row (in place) ----------------
__global__ __launch_bounds__(256)
void softmax_row_f16(_Float16* P, int cols)
{
  __shared__ float red[256];
  _Float16* p = P + (size_t)blockIdx.x * cols;
  float m = -3.0e38f;
  for (int j = threadIdx.x; j < cols; j += 256) m = fmaxf(m, (float)p[j]);
  red[threadIdx.x] = m; __syncthreads();
  for (int s = 128; s > 0; s >>= 1) {
    if (threadIdx.x < s) red[threadIdx.x] = fmaxf(red[threadIdx.x], red[threadIdx.x + s]);
    __syncthreads();
  }
  m = red[0]; __syncthreads();
  float sum = 0.0f;
  for (int j = threadIdx.x; j < cols; j += 256) sum += __expf((float)p[j] - m);
  red[threadIdx.x] = sum; __syncthreads();
  for (int s = 128; s > 0; s >>= 1) {
    if (threadIdx.x < s) red[threadIdx.x] += red[threadIdx.x + s];
    __syncthreads();
  }
  float inv = 1.0f / red[0];
  for (int j = threadIdx.x; j < cols; j += 256)
    p[j] = (_Float16)(__expf((float)p[j] - m) * inv);
}

// ---------------- residual + LayerNorm: out = LN(x + a) ----------------
__global__ __launch_bounds__(256)
void ln_residual(const float* __restrict__ X, const float* __restrict__ A,
                 const float* __restrict__ g, const float* __restrict__ b,
                 float* __restrict__ out, int d)
{
  __shared__ float r1[256], r2[256];
  const float* x = X + (size_t)blockIdx.x * d;
  const float* a = A + (size_t)blockIdx.x * d;
  float s = 0.0f, sq = 0.0f;
  for (int j = threadIdx.x; j < d; j += 256) { float v = x[j] + a[j]; s += v; sq += v * v; }
  r1[threadIdx.x] = s; r2[threadIdx.x] = sq; __syncthreads();
  for (int st = 128; st > 0; st >>= 1) {
    if (threadIdx.x < st) { r1[threadIdx.x] += r1[threadIdx.x + st]; r2[threadIdx.x] += r2[threadIdx.x + st]; }
    __syncthreads();
  }
  float mean = r1[0] / d;
  float var  = r2[0] / d - mean * mean;
  float rs   = rsqrtf(var + 1e-5f);
  float* o = out + (size_t)blockIdx.x * d;
  for (int j = threadIdx.x; j < d; j += 256)
    o[j] = (x[j] + a[j] - mean) * rs * g[j] + b[j];
}

// ---------------- graph / scatter kernels ----------------
__global__ void k_count_edges(const int* __restrict__ dst, float* cnt) {
  int e = blockIdx.x * blockDim.x + threadIdx.x;
  if (e < NEDGE) atomicAdd(&cnt[dst[e]], 1.0f);
}
__global__ void k_loopsum(const float* __restrict__ eemb, const int* __restrict__ dst,
                          float* loopsum) {
  int i = blockIdx.x * blockDim.x + threadIdx.x;
  if (i < NEDGE * DIN) {
    int e = i / DIN, d = i % DIN;
    atomicAdd(&loopsum[dst[e] * DIN + d], eemb[i]);
  }
}
__global__ void k_copy_f32(const float* __restrict__ a, float* __restrict__ b, int n) {
  int i = blockIdx.x * blockDim.x + threadIdx.x;
  if (i < n) b[i] = a[i];
}
__global__ void k_loop_build(const float* __restrict__ loopsum, const float* __restrict__ cnt,
                             float* __restrict__ eaTail) {
  int i = blockIdx.x * blockDim.x + threadIdx.x;
  if (i < NATOMS * DIN) eaTail[i] = loopsum[i] / fmaxf(cnt[i / DIN], 1.0f);
}
// fold att_edge into lin_edge: W2[c,h] = sum_d w_edge[c, h*78+d] * att_edge[h,d]
__global__ void k_w2(const float* __restrict__ w_edge, const float* __restrict__ att_edge,
                     float* __restrict__ W2) {
  int i = blockIdx.x * blockDim.x + threadIdx.x;
  if (i < DIN * NH) {
    int c = i / NH, h = i % NH;
    float sum = 0.0f;
    for (int d = 0; d < DIN; ++d) sum += w_edge[c * DH + h * DIN + d] * att_edge[h * DIN + d];
    W2[c * NH + h] = sum;
  }
}
__global__ void k_attn_sd(const float* __restrict__ xw, const float* __restrict__ att_src,
                          const float* __restrict__ att_dst,
                          float* __restrict__ as_, float* __restrict__ ad_) {
  int i = blockIdx.x * blockDim.x + threadIdx.x;
  if (i < NATOMS * NH) {
    int node = i / NH, h = i % NH;
    const float* r = xw + (size_t)node * DH + h * DIN;
    float s1 = 0.0f, s2 = 0.0f;
    for (int d = 0; d < DIN; ++d) { s1 += r[d] * att_src[h * DIN + d]; s2 += r[d] * att_dst[h * DIN + d]; }
    as_[i] = s1; ad_[i] = s2;
  }
}
__global__ void k_gat_alpha(const float* __restrict__ as_, const float* __restrict__ ad_,
                            const float* __restrict__ ae, const int* __restrict__ src,
                            const int* __restrict__ dst, float* __restrict__ alpha,
                            unsigned* __restrict__ amaxk) {
  int i = blockIdx.x * blockDim.x + threadIdx.x;
  if (i < NSELF * NH) {
    int e = i / NH, h = i % NH;
    int s, d2;
    if (e < NEDGE) { s = src[e]; d2 = dst[e]; } else { s = d2 = e - NEDGE; }
    float al = as_[s * NH + h] + ad_[d2 * NH + h] + ae[e * NH + h];
    al = al > 0.0f ? al : 0.2f * al;         // leaky_relu(0.2)
    alpha[i] = al;
    atomicMax(&amaxk[d2 * NH + h], f2k(al));
  }
}
__global__ void k_gat_exp(float* __restrict__ alpha, const unsigned* __restrict__ amaxk,
                          float* __restrict__ asum, const int* __restrict__ dst) {
  int i = blockIdx.x * blockDim.x + threadIdx.x;
  if (i < NSELF * NH) {
    int e = i / NH, h = i % NH;
    int d2 = e < NEDGE ? dst[e] : e - NEDGE;
    float ex = __expf(alpha[i] - k2f(amaxk[d2 * NH + h]));
    alpha[i] = ex;
    atomicAdd(&asum[d2 * NH + h], ex);
  }
}
__global__ void k_gat_scatter(const float* __restrict__ ex, const float* __restrict__ asum,
                              const float* __restrict__ xw, const int* __restrict__ src,
                              const int* __restrict__ dst, float* __restrict__ out) {
  int i = blockIdx.x * blockDim.x + threadIdx.x;
  if (i < NSELF * NH) {
    int e = i / NH, h = i % NH;
    int s  = e < NEDGE ? src[e] : e - NEDGE;
    int d2 = e < NEDGE ? dst[e] : e - NEDGE;
    float att = ex[i] / (asum[d2 * NH + h] + 1e-16f);
    const float* xr = xw + (size_t)s * DH + h * DIN;
    float* orow = out + (size_t)d2 * DH + h * DIN;
    for (int j = 0; j < DIN; ++j) atomicAdd(&orow[j], att * xr[j]);
  }
}
__global__ void k_bias_relu(float* __restrict__ buf, const float* __restrict__ bias, int n, int d) {
  int i = blockIdx.x * blockDim.x + threadIdx.x;
  if (i < n * d) {
    float v = buf[i] + bias[i % d];
    buf[i] = v > 0.0f ? v : 0.0f;
  }
}
__global__ void k_deg(const int* __restrict__ dst, float* deg) {
  int i = blockIdx.x * blockDim.x + threadIdx.x;
  if (i < NSELF) atomicAdd(&deg[i < NEDGE ? dst[i] : i - NEDGE], 1.0f);
}
__global__ void k_dis(float* deg) {
  int i = blockIdx.x * blockDim.x + threadIdx.x;
  if (i < NATOMS) deg[i] = rsqrtf(fmaxf(deg[i], 1.0f));
}
__global__ void k_gcn_scatter(const float* __restrict__ xw, const float* __restrict__ dis,
                              const int* __restrict__ src, const int* __restrict__ dst,
                              float* __restrict__ out) {
  int i = blockIdx.x * blockDim.x + threadIdx.x;
  if (i < NSELF * DH) {
    int e = i / DH, j = i % DH;
    int s  = e < NEDGE ? src[e] : e - NEDGE;
    int d2 = e < NEDGE ? dst[e] : e - NEDGE;
    atomicAdd(&out[(size_t)d2 * DH + j], xw[(size_t)s * DH + j] * dis[s] * dis[d2]);
  }
}
__global__ void k_pool_cnt(const int* __restrict__ batch, float* gcnt) {
  int i = blockIdx.x * blockDim.x + threadIdx.x;
  if (i < NATOMS) atomicAdd(&gcnt[batch[i]], 1.0f);
}
__global__ void k_pool_acc(const float* __restrict__ h, const int* __restrict__ batch,
                           unsigned* __restrict__ gmpk, float* __restrict__ gaps) {
  int i = blockIdx.x * blockDim.x + threadIdx.x;
  if (i < NATOMS * DH) {
    int node = i / DH, j = i % DH;
    int g = batch[node];
    atomicMax(&gmpk[g * DH + j], f2k(h[i]));
    atomicAdd(&gaps[(size_t)g * DH + j], h[i]);
  }
}
__global__ void k_pool_z(const unsigned* __restrict__ gmpk, const float* __restrict__ gaps,
                         const float* __restrict__ gcnt, float* __restrict__ z) {
  int i = blockIdx.x * blockDim.x + threadIdx.x;
  if (i < NG * DH) {
    int g = i / DH, j = i % DH;
    float c = gcnt[g];
    z[(size_t)g * (2 * DH) + j]      = c > 0.0f ? k2f(gmpk[g * DH + j]) : 0.0f;
    z[(size_t)g * (2 * DH) + DH + j] = gaps[i] / fmaxf(c, 1.0f);
  }
}

// ---------------- host side ----------------
template<bool OH, bool RELU, bool BIAS>
static void launch_gemm(const _Float16* A, const _Float16* B, const float* bias, void* C,
                        int M, int N, int K, int lda, int ldb, int ldc, float alpha,
                        hipStream_t s)
{
  dim3 grid(CDIV(N, 128), CDIV(M, 64)), blk(128);
  gemm_wmma<OH, RELU, BIAS><<<grid, blk, 0, s>>>(A, B, bias, C, M, N, K, lda, ldb, ldc, alpha);
}
static void cvtA(const float* src, _Float16* dst, int R, int C, int Cpad, hipStream_t s) {
  k_cvtA<<<CDIV(R * Cpad, 256), 256, 0, s>>>(src, dst, R, C, Cpad);
}
static void cvtT(const float* src, _Float16* dst, int R, int C, int Rpad, int Cpad, hipStream_t s) {
  k_cvtT<<<CDIV(Cpad * Rpad, 256), 256, 0, s>>>(src, dst, R, C, Rpad, Cpad);
}
static inline int pad32(int x)  { return (x + 31) & ~31; }
static inline int pad128(int x) { return (x + 127) & ~127; }

struct TL {
  const float *wq, *bq, *wk, *bk, *wv, *bv, *wo, *bo;
  const float *g1, *be1, *w1, *bf1, *w2, *bf2, *g2, *be2;
};

extern "C" void kernel_launch(void* const* d_in, const int* in_sizes, int n_in,
                              void* d_out, int out_size, void* d_ws, size_t ws_size,
                              hipStream_t stream)
{
  (void)in_sizes; (void)n_in; (void)out_size; (void)ws_size;
  const float* x      = (const float*)d_in[0];
  const int*   eidx   = (const int*)d_in[1];
  const float* eattr  = (const float*)d_in[2];
  const int*   batch  = (const int*)d_in[3];
  const float* edge_w = (const float*)d_in[4];
  const float* edge_b = (const float*)d_in[5];
  auto tl = [&](int b) {
    TL p;
    p.wq = (const float*)d_in[b + 0];  p.bq  = (const float*)d_in[b + 1];
    p.wk = (const float*)d_in[b + 2];  p.bk  = (const float*)d_in[b + 3];
    p.wv = (const float*)d_in[b + 4];  p.bv  = (const float*)d_in[b + 5];
    p.wo = (const float*)d_in[b + 6];  p.bo  = (const float*)d_in[b + 7];
    p.g1 = (const float*)d_in[b + 8];  p.be1 = (const float*)d_in[b + 9];
    p.w1 = (const float*)d_in[b + 10]; p.bf1 = (const float*)d_in[b + 11];
    p.w2 = (const float*)d_in[b + 12]; p.bf2 = (const float*)d_in[b + 13];
    p.g2 = (const float*)d_in[b + 14]; p.be2 = (const float*)d_in[b + 15];
    return p;
  };
  TL t1 = tl(6), t2 = tl(28);
  const float* gat_w    = (const float*)d_in[22];
  const float* gat_we   = (const float*)d_in[23];
  const float* att_src  = (const float*)d_in[24];
  const float* att_dst  = (const float*)d_in[25];
  const float* att_edge = (const float*)d_in[26];
  const float* gat_b    = (const float*)d_in[27];
  const float* gcn_w    = (const float*)d_in[44];
  const float* gcn_b    = (const float*)d_in[45];
  const float* fc1_w    = (const float*)d_in[46];
  const float* fc1_b    = (const float*)d_in[47];
  const float* fc2_w    = (const float*)d_in[48];
  const float* fc2_b    = (const float*)d_in[49];
  const int* esrc = eidx;
  const int* edst = eidx + NEDGE;
  float* out = (float*)d_out;

  // ---- workspace carve-up ----
  char* ws = (char*)d_ws;
  size_t off = 0;
  auto alloc = [&](size_t bytes) { char* p = ws + off; off += (bytes + 255) & ~(size_t)255; return p; };
  const int P32  = pad32(DH);    // 800
  const int P128 = pad128(DH);   // 896  (B-operand row padding for 128-wide N tiles)
  // f16 buffers
  _Float16* pP    = (_Float16*)alloc((size_t)NATOMS * NATOMS * 2);     // scores/probs
  _Float16* hIN   = (_Float16*)alloc((size_t)NATOMS * P32 * 2);
  _Float16* hQ    = (_Float16*)alloc((size_t)NATOMS * P32 * 2);
  _Float16* hK    = (_Float16*)alloc((size_t)NATOMS * P32 * 2);
  _Float16* hVT   = (_Float16*)alloc((size_t)P128 * NATOMS * 2);
  _Float16* hO    = (_Float16*)alloc((size_t)NATOMS * P32 * 2);
  _Float16* hX1   = (_Float16*)alloc((size_t)NATOMS * P32 * 2);
  _Float16* hF1   = (_Float16*)alloc((size_t)NATOMS * DFF * 2);
  _Float16* hEA   = (_Float16*)alloc((size_t)NSELF * pad32(DIN) * 2);
  _Float16* hEAT  = (_Float16*)alloc((size_t)NEDGE * 32 * 2);          // edge_attr padded
  _Float16* hZ    = (_Float16*)alloc((size_t)NG * pad32(2 * DH) * 2);
  _Float16* hZF   = (_Float16*)alloc((size_t)NG * DFC1 * 2);
  _Float16* wT0   = (_Float16*)alloc((size_t)P128 * P32 * 2);
  _Float16* wT1   = (_Float16*)alloc((size_t)P128 * P32 * 2);
  _Float16* wT2   = (_Float16*)alloc((size_t)P128 * P32 * 2);
  _Float16* wT3   = (_Float16*)alloc((size_t)P128 * P32 * 2);
  _Float16* w1T   = (_Float16*)alloc((size_t)DFF * P32 * 2);
  _Float16* w2T   = (_Float16*)alloc((size_t)P128 * DFF * 2);
  _Float16* ewT   = (_Float16*)alloc((size_t)128 * 32 * 2);
  _Float16* gatwT = (_Float16*)alloc((size_t)P128 * pad32(DIN) * 2);
  _Float16* w2fT  = (_Float16*)alloc((size_t)128 * pad32(DIN) * 2);    // folded edge-att
  _Float16* gcnwT = (_Float16*)alloc((size_t)P128 * P32 * 2);
  _Float16* fc1T  = (_Float16*)alloc((size_t)DFC1 * pad32(2 * DH) * 2);
  _Float16* fc2T  = (_Float16*)alloc((size_t)DFC2 * DFC1 * 2);
  // fp32 buffers
  float* pV    = (float*)alloc((size_t)NATOMS * DH * 4);
  float* pT    = (float*)alloc((size_t)NATOMS * DH * 4);
  float* pX1   = (float*)alloc((size_t)NATOMS * DH * 4);
  float* pX2   = (float*)alloc((size_t)NATOMS * DH * 4);
  float* pH    = (float*)alloc((size_t)NATOMS * DH * 4);
  float* pXW   = (float*)alloc((size_t)NATOMS * DH * 4);
  float* pEEMB = (float*)alloc((size_t)NEDGE * DIN * 4);
  float* pEA   = (float*)alloc((size_t)NSELF * DIN * 4);
  float* pAE   = (float*)alloc((size_t)NSELF * NH * 4);
  float* pALPH = (float*)alloc((size_t)NSELF * NH * 4);
  float* pAS   = (float*)alloc((size_t)NATOMS * NH * 4);
  float* pAD   = (float*)alloc((size_t)NATOMS * NH * 4);
  unsigned* pAMAX = (unsigned*)alloc((size_t)NATOMS * NH * 4);
  float* pASUM = (float*)alloc((size_t)NATOMS * NH * 4);
  float* pCNT  = (float*)alloc((size_t)NATOMS * 4);
  float* pLSUM = (float*)alloc((size_t)NATOMS * DIN * 4);
  float* pW2   = (float*)alloc((size_t)DIN * NH * 4);
  float* pDEG  = (float*)alloc((size_t)NATOMS * 4);
  float* pGCNT = (float*)alloc((size_t)NG * 4);
  unsigned* pGMPK = (unsigned*)alloc((size_t)NG * DH * 4);
  float* pGAPS = (float*)alloc((size_t)NG * DH * 4);
  float* pZ    = (float*)alloc((size_t)NG * 2 * DH * 4);
  float* pO    = (float*)alloc((size_t)NATOMS * DH * 4);   // GCN accumulator

  // ---- one TransformerEncoderLayer (nhead=1, post-norm, relu) ----
  auto run_transformer = [&](const float* xin, const TL& p, int d, float* xout) {
    const int p32 = pad32(d), p128 = pad128(d);
    const float scale = 1.0f / sqrtf((float)d);
    // weights -> f16 trans layout (rows padded to 128 for guard-free staging)
    cvtT(p.wq, wT0, d, d, p32, p128, stream);
    cvtT(p.wk, wT1, d, d, p32, p128, stream);
    cvtT(p.wv, wT2, d, d, p32, p128, stream);
    cvtT(p.wo, wT3, d, d, p32, p128, stream);
    cvtT(p.w1, w1T, d, DFF, p32, DFF, stream);
    cvtT(p.w2, w2T, DFF, d, DFF, p128, stream);
    // input -> f16 padded
    cvtA(xin, hIN, NATOMS, d, p32, stream);
    // zero padded columns of f16 GEMM outputs
    hipMemsetAsync(hQ, 0, (size_t)NATOMS * p32 * 2, stream);
    hipMemsetAsync(hK, 0, (size_t)NATOMS * p32 * 2, stream);
    hipMemsetAsync(hO, 0, (size_t)NATOMS * p32 * 2, stream);
    // q,k (f16 direct), v (fp32, then transpose-convert)
    launch_gemm<true,  false, true>(hIN, wT0, p.bq, hQ, NATOMS, d, p32, p32, p32, p32, 1.0f, stream);
    launch_gemm<true,  false, true>(hIN, wT1, p.bk, hK, NATOMS, d, p32, p32, p32, p32, 1.0f, stream);
    launch_gemm<false, false, true>(hIN, wT2, p.bv, pV, NATOMS, d, p32, p32, p32, d,   1.0f, stream);
    // S = scale * Q @ K^T  -> f16, L2-resident
    launch_gemm<true, false, false>(hQ, hK, nullptr, pP, NATOMS, NATOMS, p32, p32, p32, NATOMS, scale, stream);
    softmax_row_f16<<<NATOMS, 256, 0, stream>>>(pP, NATOMS);
    cvtT(pV, hVT, NATOMS, d, NATOMS, p128, stream);
    // O = P @ V  (f16 out, padded cols pre-zeroed)
    launch_gemm<true, false, false>(pP, hVT, nullptr, hO, NATOMS, d, NATOMS, NATOMS, NATOMS, p32, 1.0f, stream);
    // attn projection, residual+LN
    launch_gemm<false, false, true>(hO, wT3, p.bo, pT, NATOMS, d, p32, p32, p32, d, 1.0f, stream);
    ln_residual<<<NATOMS, 256, 0, stream>>>(xin, pT, p.g1, p.be1, pX1, d);
    cvtA(pX1, hX1, NATOMS, d, p32, stream);
    // FFN
    launch_gemm<true,  true,  true>(hX1, w1T, p.bf1, hF1, NATOMS, DFF, p32, p32, p32, DFF, 1.0f, stream);
    launch_gemm<false, false, true>(hF1, w2T, p.bf2, pT, NATOMS, d, DFF, DFF, DFF, d, 1.0f, stream);
    ln_residual<<<NATOMS, 256, 0, stream>>>(pX1, pT, p.g2, p.be2, xout, d);
  };

  // ================= pipeline =================
  // edge embedding: [E,10] @ [10,78] + b
  cvtA(eattr, hEAT, NEDGE, 10, 32, stream);
  cvtT(edge_w, ewT, 10, DIN, 32, 128, stream);
  launch_gemm<false, false, true>(hEAT, ewT, edge_b, pEEMB, NEDGE, DIN, 32, 32, 32, DIN, 1.0f, stream);

  // transformer 1 on raw atom features (d=78)
  run_transformer(x, t1, DIN, pX2);

  // ---- GAT (heads=10, edge_dim=78, self loops w/ mean edge_attr) ----
  hipMemsetAsync(pCNT, 0, (size_t)NATOMS * 4, stream);
  hipMemsetAsync(pLSUM, 0, (size_t)NATOMS * DIN * 4, stream);
  k_count_edges<<<CDIV(NEDGE, 256), 256, 0, stream>>>(edst, pCNT);
  k_loopsum<<<CDIV(NEDGE * DIN, 256), 256, 0, stream>>>(pEEMB, edst, pLSUM);
  k_copy_f32<<<CDIV(NEDGE * DIN, 256), 256, 0, stream>>>(pEEMB, pEA, NEDGE * DIN);
  k_loop_build<<<CDIV(NATOMS * DIN, 256), 256, 0, stream>>>(pLSUM, pCNT, pEA + (size_t)NEDGE * DIN);
  k_w2<<<CDIV(DIN * NH, 256), 256, 0, stream>>>(gat_we, att_edge, pW2);
  cvtA(pEA, hEA, NSELF, DIN, pad32(DIN), stream);
  cvtT(pW2, w2fT, DIN, NH, pad32(DIN), 128, stream);
  launch_gemm<false, false, false>(hEA, w2fT, nullptr, pAE, NSELF, NH, pad32(DIN),
                                   pad32(DIN), pad32(DIN), NH, 1.0f, stream);
  cvtA(pX2, hIN, NATOMS, DIN, pad32(DIN), stream);
  cvtT(gat_w, gatwT, DIN, DH, pad32(DIN), P128, stream);
  launch_gemm<false, false, false>(hIN, gatwT, nullptr, pXW, NATOMS, DH, pad32(DIN),
                                   pad32(DIN), pad32(DIN), DH, 1.0f, stream);
  k_attn_sd<<<CDIV(NATOMS * NH, 256), 256, 0, stream>>>(pXW, att_src, att_dst, pAS, pAD);
  hipMemsetAsync(pAMAX, 0, (size_t)NATOMS * NH * 4, stream);   // key 0 == -inf
  hipMemsetAsync(pASUM, 0, (size_t)NATOMS * NH * 4, stream);
  k_gat_alpha<<<CDIV(NSELF * NH, 256), 256, 0, stream>>>(pAS, pAD, pAE, esrc, edst, pALPH, pAMAX);
  k_gat_exp<<<CDIV(NSELF * NH, 256), 256, 0, stream>>>(pALPH, pAMAX, pASUM, edst);
  hipMemsetAsync(pH, 0, (size_t)NATOMS * DH * 4, stream);
  k_gat_scatter<<<CDIV(NSELF * NH, 256), 256, 0, stream>>>(pALPH, pASUM, pXW, esrc, edst, pH);
  k_bias_relu<<<CDIV(NATOMS * DH, 256), 256, 0, stream>>>(pH, gat_b, NATOMS, DH);

  // transformer 2 on GAT output (d=780) -- dominant compute
  run_transformer(pH, t2, DH, pX2);

  // ---- GCN (sym-norm, self loops) + relu ----
  cvtA(pX2, hIN, NATOMS, DH, P32, stream);
  cvtT(gcn_w, gcnwT, DH, DH, P32, P128, stream);
  launch_gemm<false, false, false>(hIN, gcnwT, nullptr, pXW, NATOMS, DH, P32, P32, P32, DH, 1.0f, stream);
  hipMemsetAsync(pDEG, 0, (size_t)NATOMS * 4, stream);
  k_deg<<<CDIV(NSELF, 256), 256, 0, stream>>>(edst, pDEG);
  k_dis<<<CDIV(NATOMS, 256), 256, 0, stream>>>(pDEG);
  hipMemsetAsync(pO, 0, (size_t)NATOMS * DH * 4, stream);
  k_gcn_scatter<<<CDIV(NSELF * DH, 256), 256, 0, stream>>>(pXW, pDEG, esrc, edst, pO);
  k_bias_relu<<<CDIV(NATOMS * DH, 256), 256, 0, stream>>>(pO, gcn_b, NATOMS, DH);

  // ---- global max+mean pool, FC head ----
  hipMemsetAsync(pGCNT, 0, (size_t)NG * 4, stream);
  hipMemsetAsync(pGMPK, 0, (size_t)NG * DH * 4, stream);
  hipMemsetAsync(pGAPS, 0, (size_t)NG * DH * 4, stream);
  k_pool_cnt<<<CDIV(NATOMS, 256), 256, 0, stream>>>(batch, pGCNT);
  k_pool_acc<<<CDIV(NATOMS * DH, 256), 256, 0, stream>>>(pO, batch, pGMPK, pGAPS);
  k_pool_z<<<CDIV(NG * DH, 256), 256, 0, stream>>>(pGMPK, pGAPS, pGCNT, pZ);
  cvtA(pZ, hZ, NG, 2 * DH, pad32(2 * DH), stream);
  cvtT(fc1_w, fc1T, 2 * DH, DFC1, pad32(2 * DH), DFC1, stream);
  cvtT(fc2_w, fc2T, DFC1, DFC2, DFC1, DFC2, stream);
  launch_gemm<true, true, true>(hZ, fc1T, fc1_b, hZF, NG, DFC1, pad32(2 * DH),
                                pad32(2 * DH), pad32(2 * DH), DFC1, 1.0f, stream);
  launch_gemm<false, false, true>(hZF, fc2T, fc2_b, out, NG, DFC2, DFC1, DFC1, DFC1, DFC2, 1.0f, stream);
}
